// MotionFeatureEncoder_27986006901436
// MI455X (gfx1250) — compile-verified, matrix operand order checked by reading, not run on hardware
//
#include <hip/hip_runtime.h>

// Mamba motion-feature encoder for gfx1250 (MI455X).
// f32 weights/activations pre-packed to zero-padded f16; GEMMs double-buffer
// tiles in LDS via GLOBAL_LOAD_ASYNC_TO_LDS_B128 (ASYNCcnt) and consume them
// with v_wmma_f32_16x16x32_f16. Fused conv+silu, register-resident scan, LN.

#define Bsz    64
#define Lseq   512
#define DMODEL 256
#define DINNER 512
#define DSTATE 16
#define DCONV  4
#define DTRANK 16
#define NLAYERS 2
#define INDIM  41
#define Mrows  (Bsz * Lseq)   // 32768 token rows

typedef __attribute__((ext_vector_type(16))) _Float16 v16h;
typedef __attribute__((ext_vector_type(8)))  _Float16 v8h;
typedef __attribute__((ext_vector_type(8)))  float    v8f;

__device__ __forceinline__ float silu_f(float x) { return x / (1.f + __expf(-x)); }
__device__ __forceinline__ float softplus_f(float x) {
  return x > 20.f ? x : __logf(1.f + __expf(x));
}

// ---------------------------------------------------------------------------
// Concatenate [speed|bbox|pose] -> motion [Mrows, 41]
// ---------------------------------------------------------------------------
__global__ void concat_motion(const float* __restrict__ sp,
                              const float* __restrict__ bb,
                              const float* __restrict__ ps,
                              float* __restrict__ motion) {
  int idx = blockIdx.x * blockDim.x + threadIdx.x;
  if (idx >= Mrows * INDIM) return;
  int bl = idx / INDIM, j = idx % INDIM;
  float v;
  if (j == 0)      v = sp[bl];
  else if (j < 5)  v = bb[bl * 4 + (j - 1)];
  else             v = ps[bl * 36 + (j - 5)];
  motion[idx] = v;
}

// ---------------------------------------------------------------------------
// Pack f32 [Rsrc, C] (row stride ld) -> f16 [R, Cp] zero-padded.
// ---------------------------------------------------------------------------
__global__ void pack_f16(const float* __restrict__ src, int ld, int Rsrc,
                         int C, int Cp, int R, _Float16* __restrict__ dst) {
  int idx = blockIdx.x * blockDim.x + threadIdx.x;
  if (idx >= R * Cp) return;
  int r = idx / Cp, c = idx % Cp;
  float v = (r < Rsrc && c < C) ? src[(size_t)r * ld + c] : 0.f;
  dst[idx] = (_Float16)v;
}

// ---------------------------------------------------------------------------
// WMMA GEMM on packed f16: C[M,N] = A[M,Kp] * W[Np,Kp]^T  (+ epilogue)
//   mode 0: plain   mode 1: softplus(acc + bias[col])   mode 2: acc + res
// Block: 256 threads = 8 waves, tile 64(M) x 64(N), K stepped by 32.
// Wave grid 4x2; each wave computes a 16x32 slab = two v_wmma accumulators
// sharing one A fragment. Tiles are double-buffered in LDS and filled with
// global_load_async_to_lds_b128 (one async b128 per thread per matrix),
// overlapping the next tile's fill with the current tile's WMMA work.
// One barrier per K-step: s_wait_asynccnt 0 + workgroup barrier.
// ---------------------------------------------------------------------------
__global__ __launch_bounds__(256)
void gemm_wmma(const _Float16* __restrict__ A,
               const _Float16* __restrict__ W,
               float* __restrict__ C, int ldc,
               const float* __restrict__ bias,
               const float* __restrict__ res, int ldres,
               int N, int Kp, int mode) {
  __shared__ _Float16 As[2][64][40];   // 80B row stride keeps b128 alignment
  __shared__ _Float16 Ws[2][64][40];

  const int t    = threadIdx.x;
  const int lane = t & 31;
  const int wave = t >> 5;
  const int wm   = wave >> 1;       // 0..3 (M sub-tile)
  const int wn   = wave & 1;        // 0..1 (N slab of 32)
  const int m0   = blockIdx.y * 64;
  const int n0   = blockIdx.x * 64;

  const int lrow = t >> 2;          // 0..63
  const int lcol = (t & 3) * 8;     // 0,8,16,24

  v8f acc0 = {}, acc1 = {};
  const int ksteps = Kp >> 5;

  // async fill of tile kk into LDS buffer `buf` (per-thread b128, no VGPR data)
  auto issue_tile = [&](int kk, int buf) {
    const _Float16* asrc = A + (size_t)(m0 + lrow) * Kp + kk * 32 + lcol;
    const _Float16* wsrc = W + (size_t)(n0 + lrow) * Kp + kk * 32 + lcol;
    unsigned la = (unsigned)(size_t)&As[buf][lrow][lcol];
    unsigned lw = (unsigned)(size_t)&Ws[buf][lrow][lcol];
    asm volatile("global_load_async_to_lds_b128 %0, %1, off"
                 :: "v"(la), "v"(asrc) : "memory");
    asm volatile("global_load_async_to_lds_b128 %0, %1, off"
                 :: "v"(lw), "v"(wsrc) : "memory");
  };

  issue_tile(0, 0);
  for (int kk = 0; kk < ksteps; ++kk) {
    const int buf = kk & 1;
    asm volatile("s_wait_asynccnt 0" ::: "memory");  // this wave's fills landed
    __syncthreads();                                 // all waves' fills landed
    if (kk + 1 < ksteps) issue_tile(kk + 1, buf ^ 1);

    // gather fragments per ISA 16-bit 16x32 layout (merges to ds_load_b128)
    v16h af, bf0, bf1;
    const int mrow  = wm * 16 + (lane & 15);
    const int nrow0 = wn * 32 + (lane & 15);
    const int half  = lane >> 4;
#pragma unroll
    for (int i = 0; i < 8; ++i) {
      int kb = ((i < 4) ? 0 : 16) + 8 * half + 2 * (i & 3);
      af[2 * i]      = As[buf][mrow][kb];
      af[2 * i + 1]  = As[buf][mrow][kb + 1];
      bf0[2 * i]     = Ws[buf][nrow0][kb];
      bf0[2 * i + 1] = Ws[buf][nrow0][kb + 1];
      bf1[2 * i]     = Ws[buf][nrow0 + 16][kb];
      bf1[2 * i + 1] = Ws[buf][nrow0 + 16][kb + 1];
    }
    acc0 = __builtin_amdgcn_wmma_f32_16x16x32_f16(false, af, false, bf0,
                                                  (short)0, acc0, false, false);
    acc1 = __builtin_amdgcn_wmma_f32_16x16x32_f16(false, af, false, bf1,
                                                  (short)0, acc1, false, false);
  }

  // store D: VGPR r -> row = 8*half + r ; col = lane&15 (+16 for acc1)
  const int half = lane >> 4;
  const int col0 = n0 + wn * 32 + (lane & 15);
#pragma unroll
  for (int r = 0; r < 8; ++r) {
    const int row = m0 + wm * 16 + half * 8 + r;
    float v0 = acc0[r], v1 = acc1[r];
    if (mode == 1) {
      if (col0 < N)      v0 = softplus_f(v0 + bias[col0]);
      if (col0 + 16 < N) v1 = softplus_f(v1 + bias[col0 + 16]);
    } else if (mode == 2) {
      if (col0 < N)      v0 += res[(size_t)row * ldres + col0];
      if (col0 + 16 < N) v1 += res[(size_t)row * ldres + col0 + 16];
    }
    if (col0 < N)      C[(size_t)row * ldc + col0]      = v0;
    if (col0 + 16 < N) C[(size_t)row * ldc + col0 + 16] = v1;
  }
}

// ---------------------------------------------------------------------------
// LayerNorm over last dim (=256). One block (256 threads) per row.
// ---------------------------------------------------------------------------
__global__ __launch_bounds__(256)
void layernorm_256(const float* __restrict__ x, const float* __restrict__ scale,
                   const float* __restrict__ bias, float* __restrict__ y) {
  __shared__ float red[8];
  const int row = blockIdx.x;
  const int t = threadIdx.x;
  float v = x[(size_t)row * DMODEL + t];

  float s = v;
  for (int o = 16; o > 0; o >>= 1) s += __shfl_xor(s, o, 32);
  if ((t & 31) == 0) red[t >> 5] = s;
  __syncthreads();
  if (t < 8) {
    float q = red[t];
    for (int o = 4; o > 0; o >>= 1) q += __shfl_xor(q, o, 8);
    if (t == 0) red[0] = q;
  }
  __syncthreads();
  const float mu = red[0] / (float)DMODEL;
  __syncthreads();

  const float d = v - mu;
  float s2 = d * d;
  for (int o = 16; o > 0; o >>= 1) s2 += __shfl_xor(s2, o, 32);
  if ((t & 31) == 0) red[t >> 5] = s2;
  __syncthreads();
  if (t < 8) {
    float q = red[t];
    for (int o = 4; o > 0; o >>= 1) q += __shfl_xor(q, o, 8);
    if (t == 0) red[0] = q;
  }
  __syncthreads();
  const float var = red[0] / (float)DMODEL;
  y[(size_t)row * DMODEL + t] = d * rsqrtf(var + 1e-5f) * scale[t] + bias[t];
}

// ---------------------------------------------------------------------------
// Causal depthwise conv1d (kernel 4) + SiLU.  xin = xz[:, 0:512].
// ---------------------------------------------------------------------------
__global__ __launch_bounds__(256)
void conv_silu(const float* __restrict__ xz, const float* __restrict__ cw,
               const float* __restrict__ cb, float* __restrict__ u) {
  int idx = blockIdx.x * blockDim.x + threadIdx.x;   // Mrows * DINNER
  if (idx >= Mrows * DINNER) return;
  int d  = idx & (DINNER - 1);
  int bl = idx >> 9;
  int l  = bl & (Lseq - 1);
  int b  = bl >> 9;
  float acc = cb[d];
#pragma unroll
  for (int j = 0; j < DCONV; ++j) {
    int lt = l - (DCONV - 1) + j;
    if (lt >= 0)
      acc += xz[(size_t)(b * Lseq + lt) * (2 * DINNER) + d] * cw[d * DCONV + j];
  }
  u[idx] = silu_f(acc);
}

// ---------------------------------------------------------------------------
// Selective scan: one thread per (b, d); 16-state in registers; serial over L.
// Fuses +u*D and *silu(z) gating; writes yg.
// ---------------------------------------------------------------------------
__global__ __launch_bounds__(256)
void ssm_scan(const float* __restrict__ delta, const float* __restrict__ u,
              const float* __restrict__ xdbl, const float* __restrict__ xz,
              const float* __restrict__ A_log, const float* __restrict__ Dp,
              float* __restrict__ yg) {
  int idx = blockIdx.x * blockDim.x + threadIdx.x;   // Bsz * DINNER
  if (idx >= Bsz * DINNER) return;
  const int d = idx & (DINNER - 1);
  const int b = idx >> 9;

  float Arow[DSTATE];
#pragma unroll
  for (int s = 0; s < DSTATE; ++s) Arow[s] = -__expf(A_log[d * DSTATE + s]);
  const float Dv = Dp[d];

  float h[DSTATE];
#pragma unroll
  for (int s = 0; s < DSTATE; ++s) h[s] = 0.f;

  for (int t = 0; t < Lseq; ++t) {
    const size_t base = (size_t)(b * Lseq + t);
    const float dt = delta[base * DINNER + d];
    const float ut = u[base * DINNER + d];
    const float zt = xz[base * (2 * DINNER) + DINNER + d];
    const float* xd = xdbl + base * 48;    // [dt_r(16) | Bc(16) | Cc(16)]
    const float du = dt * ut;
    float y = 0.f;
#pragma unroll
    for (int s = 0; s < DSTATE; ++s) {
      h[s] = __expf(dt * Arow[s]) * h[s] + du * xd[16 + s];
      y += h[s] * xd[32 + s];
    }
    y += ut * Dv;
    yg[base * DINNER + d] = y * silu_f(zt);
  }
}

// ---------------------------------------------------------------------------
extern "C" void kernel_launch(void* const* d_in, const int* in_sizes, int n_in,
                              void* d_out, int out_size, void* d_ws, size_t ws_size,
                              hipStream_t stream) {
  (void)in_sizes; (void)n_in; (void)out_size; (void)ws_size;
  const float* speed     = (const float*)d_in[0];
  const float* bbox      = (const float*)d_in[1];
  const float* pose      = (const float*)d_in[2];
  const float* embed_w   = (const float*)d_in[3];
  const float* en_scale  = (const float*)d_in[4];
  const float* en_bias   = (const float*)d_in[5];
  const float* in_proj_w = (const float*)d_in[6];
  const float* conv_w    = (const float*)d_in[7];
  const float* conv_b    = (const float*)d_in[8];
  const float* x_proj_w  = (const float*)d_in[9];
  const float* dt_proj_w = (const float*)d_in[10];
  const float* dt_proj_b = (const float*)d_in[11];
  const float* A_log     = (const float*)d_in[12];
  const float* Dp        = (const float*)d_in[13];
  const float* out_proj_w= (const float*)d_in[14];
  const float* on_scale  = (const float*)d_in[15];
  const float* on_bias   = (const float*)d_in[16];

  float* ws = (float*)d_ws;
  size_t off = 0;
  auto alloc_f32 = [&](size_t n) { float* p = ws + off; off += (n + 3) & ~(size_t)3; return p; };
  float* motion = alloc_f32((size_t)Mrows * INDIM);
  float* x0     = alloc_f32((size_t)Mrows * DMODEL);
  float* x1     = alloc_f32((size_t)Mrows * DMODEL);
  float* xe     = alloc_f32((size_t)Mrows * DMODEL);
  float* xz     = alloc_f32((size_t)Mrows * 2 * DINNER);
  float* u      = alloc_f32((size_t)Mrows * DINNER);
  float* xdbl   = alloc_f32((size_t)Mrows * 48);
  float* delta  = alloc_f32((size_t)Mrows * DINNER);
  float* yg     = alloc_f32((size_t)Mrows * DINNER);
  _Float16* aP  = (_Float16*)alloc_f32((size_t)Mrows * DINNER / 2);  // M x 512 f16
  _Float16* wP  = (_Float16*)alloc_f32((size_t)1024 * DMODEL / 2);   // 1024 x 256 f16

  const dim3 blk(256);
  auto gpack = [](size_t n) { return (unsigned)((n + 255) / 256); };

  concat_motion<<<gpack((size_t)Mrows * INDIM), blk, 0, stream>>>(speed, bbox,
                                                                  pose, motion);
  // ---- embed: xe = motion [M,41] x embed_w[256,41]^T  (Kp=64) ----
  pack_f16<<<gpack((size_t)Mrows * 64), blk, 0, stream>>>(
      motion, INDIM, Mrows, INDIM, 64, Mrows, aP);
  pack_f16<<<gpack((size_t)DMODEL * 64), blk, 0, stream>>>(
      embed_w, INDIM, DMODEL, INDIM, 64, DMODEL, wP);
  gemm_wmma<<<dim3(DMODEL / 64, Mrows / 64), blk, 0, stream>>>(
      aP, wP, xe, DMODEL, nullptr, nullptr, 0, DMODEL, 64, 0);
  layernorm_256<<<Mrows, DMODEL, 0, stream>>>(xe, en_scale, en_bias, x0);

  float* xcur = x0;
  float* xnext = x1;
  for (int l = 0; l < NLAYERS; ++l) {
    const float* iw  = in_proj_w  + (size_t)l * 2 * DINNER * DMODEL;
    const float* cw  = conv_w     + (size_t)l * DINNER * DCONV;
    const float* cb  = conv_b     + (size_t)l * DINNER;
    const float* xpw = x_proj_w   + (size_t)l * 48 * DINNER;
    const float* dtw = dt_proj_w  + (size_t)l * DINNER * DTRANK;
    const float* dtb = dt_proj_b  + (size_t)l * DINNER;
    const float* Al  = A_log      + (size_t)l * DINNER * DSTATE;
    const float* Dl  = Dp         + (size_t)l * DINNER;
    const float* ow  = out_proj_w + (size_t)l * DMODEL * DINNER;

    // xz = x @ in_w^T  [M, 1024]  (Kp=256)
    pack_f16<<<gpack((size_t)Mrows * DMODEL), blk, 0, stream>>>(
        xcur, DMODEL, Mrows, DMODEL, DMODEL, Mrows, aP);
    pack_f16<<<gpack((size_t)1024 * DMODEL), blk, 0, stream>>>(
        iw, DMODEL, 1024, DMODEL, DMODEL, 1024, wP);
    gemm_wmma<<<dim3(1024 / 64, Mrows / 64), blk, 0, stream>>>(
        aP, wP, xz, 2 * DINNER, nullptr, nullptr, 0, 2 * DINNER, DMODEL, 0);

    // u = silu(causal depthwise conv(xin) + cb)
    conv_silu<<<gpack((size_t)Mrows * DINNER), blk, 0, stream>>>(xz, cw, cb, u);

    // x_dbl = u @ xpw^T  [M, 48]  (Kp=512, Np=64)
    pack_f16<<<gpack((size_t)Mrows * DINNER), blk, 0, stream>>>(
        u, DINNER, Mrows, DINNER, DINNER, Mrows, aP);
    pack_f16<<<gpack((size_t)64 * DINNER), blk, 0, stream>>>(
        xpw, DINNER, 48, DINNER, DINNER, 64, wP);
    gemm_wmma<<<dim3(1, Mrows / 64), blk, 0, stream>>>(
        aP, wP, xdbl, 48, nullptr, nullptr, 0, 48, DINNER, 0);

    // delta = softplus(dt_r @ dtw^T + dtb)  [M, 512]  (Kp=32)
    pack_f16<<<gpack((size_t)Mrows * 32), blk, 0, stream>>>(
        xdbl, 48, Mrows, DTRANK, 32, Mrows, aP);
    pack_f16<<<gpack((size_t)DINNER * 32), blk, 0, stream>>>(
        dtw, DTRANK, DINNER, DTRANK, 32, DINNER, wP);
    gemm_wmma<<<dim3(DINNER / 64, Mrows / 64), blk, 0, stream>>>(
        aP, wP, delta, DINNER, dtb, nullptr, 0, DINNER, 32, 1);

    // selective scan + D skip + silu(z) gate
    ssm_scan<<<gpack((size_t)Bsz * DINNER), blk, 0, stream>>>(
        delta, u, xdbl, xz, Al, Dl, yg);

    // xnext = yg @ ow^T + xcur  (Kp=512)
    pack_f16<<<gpack((size_t)Mrows * DINNER), blk, 0, stream>>>(
        yg, DINNER, Mrows, DINNER, DINNER, Mrows, aP);
    pack_f16<<<gpack((size_t)DMODEL * DINNER), blk, 0, stream>>>(
        ow, DINNER, DMODEL, DINNER, DINNER, DMODEL, wP);
    gemm_wmma<<<dim3(DMODEL / 64, Mrows / 64), blk, 0, stream>>>(
        aP, wP, xnext, DMODEL, nullptr, xcur, DMODEL, DMODEL, DINNER, 2);

    float* tmp = xcur; xcur = xnext; xnext = tmp;
  }

  layernorm_256<<<Mrows, DMODEL, 0, stream>>>(xcur, on_scale, on_bias,
                                              (float*)d_out);
}
